// EdgeConv_91336774517536
// MI455X (gfx1250) — matrix operand order repaired and checked by reading, not compiled, not run on hardware
//
#include <hip/hip_runtime.h>

typedef __attribute__((ext_vector_type(16))) _Float16 v16h;
typedef __attribute__((ext_vector_type(8)))  _Float16 v8h;
typedef __attribute__((ext_vector_type(8)))  float    v8f;
typedef __attribute__((ext_vector_type(4)))  float    v4f;

#define WPB 4   // waves per block
#define PTS 2   // points per wave per iteration

// EdgeConv: N points, D=64, K=16 neighbors, H=128.
// feat (16x128) -> relu(feat@W1+b1) -> (@W2+b2) -> max over 16 rows.
__global__ __launch_bounds__(WPB * 32) void edgeconv_wmma(
    const float* __restrict__ x,
    const long long* __restrict__ ind,
    const float* __restrict__ W1,
    const float* __restrict__ b1,
    const float* __restrict__ W2,
    const float* __restrict__ b2,
    float* __restrict__ out,
    int npts)
{
    // W1^T / W2^T as f16: layout [outCol N=128][inRow K=128], K contiguous.
    __shared__ __attribute__((aligned(32))) _Float16 lw1[128 * 128];
    __shared__ __attribute__((aligned(32))) _Float16 lw2[128 * 128];
    // per-wave h1 staging: PTS x [M=16][K=128] f16
    __shared__ __attribute__((aligned(32))) _Float16 stage[WPB * PTS * 16 * 128];

    const int tid = threadIdx.x;

    for (int i = tid; i < 128 * 128; i += WPB * 32) {
        const int r = i >> 7;    // input row (K dim of GEMM)
        const int c = i & 127;   // output col (N dim)
        lw1[c * 128 + r] = (_Float16)W1[i];
        lw2[c * 128 + r] = (_Float16)W2[i];
    }
    __syncthreads();

    const int ln   = tid & 31;
    const int half = ln >> 4;      // 0: lanes 0-15, 1: lanes 16-31
    const int nidx = ln & 15;      // row (A) / column (B,C) index
    const int K0   = half * 16;    // A-operand K base (contiguous 16 K per half-wave)
    _Float16* st   = &stage[(tid >> 5) * (PTS * 2048)];

    float b1v[8], b2v[8];
#pragma unroll
    for (int t = 0; t < 8; ++t) {
        b1v[t] = b1[t * 16 + nidx];
        b2v[t] = b2[t * 16 + nidx];
    }

    const int wave  = blockIdx.x * WPB + (tid >> 5);
    const int nwave = gridDim.x * WPB;

    for (int base = wave * PTS; base < npts; base += nwave * PTS) {
        v16h a[PTS][4];
        bool valid[PTS];

#pragma unroll
        for (int p = 0; p < PTS; ++p) {
            const int n = base + p;
            valid[p] = (n < npts);
            const long long nn = valid[p] ? n : (npts - 1); // wave-uniform clamp
            // lane's neighbor row: M = nidx (both half-waves load same index)
            const long long nb = ind[nn * 16 + nidx];
            const float* xc = x + nn * 64;
            const float* xn = x + nb * 64;
            float cv[4][8], nv[4][8];
#pragma unroll
            for (int j = 0; j < 4; ++j) {
                const int f = K0 + (j & 1) * 8 + (j >> 1) * 32;     // 8-float runs at K0+{0,8,32,40}
                const v4f c0 = *(const v4f*)(xc + f);
                const v4f c1 = *(const v4f*)(xc + f + 4);
                const v4f n0 = *(const v4f*)(xn + f);
                const v4f n1 = *(const v4f*)(xn + f + 4);
#pragma unroll
                for (int i = 0; i < 4; ++i) {
                    cv[j][i] = c0[i]; cv[j][i + 4] = c1[i];
                    nv[j][i] = n0[i]; nv[j][i + 4] = n1[i];
                }
            }
            // A tiles in contiguous K order (same as B operand):
            // halfs 0-7 -> K = kb*32+K0.. ; halfs 8-15 -> K = kb*32+K0+8..
            // features 0-63 = neigh - center, 64-127 = center
#pragma unroll
            for (int i = 0; i < 8; ++i) {
                a[p][0][i]     = (_Float16)(nv[0][i] - cv[0][i]);
                a[p][0][i + 8] = (_Float16)(nv[1][i] - cv[1][i]);
                a[p][1][i]     = (_Float16)(nv[2][i] - cv[2][i]);
                a[p][1][i + 8] = (_Float16)(nv[3][i] - cv[3][i]);
                a[p][2][i]     = (_Float16)cv[0][i];
                a[p][2][i + 8] = (_Float16)cv[1][i];
                a[p][3][i]     = (_Float16)cv[2][i];
                a[p][3][i + 8] = (_Float16)cv[3][i];
            }
        }

        // ---- GEMM1: h1 = relu(feat @ W1 + b1) -> stage ----
#pragma unroll
        for (int t = 0; t < 8; ++t) {
            v8f acc[PTS];
#pragma unroll
            for (int p = 0; p < PTS; ++p) { v8f z = {}; acc[p] = z; }
#pragma unroll
            for (int kb = 0; kb < 4; ++kb) {
                // B layout: lane col = t*16+nidx, 16 consecutive K per half-wave
                const v16h bt = *(const v16h*)(
                    lw1 + ((t * 16 + nidx) * 128 + kb * 32 + half * 16));
#pragma unroll
                for (int p = 0; p < PTS; ++p)
                    acc[p] = __builtin_amdgcn_wmma_f32_16x16x32_f16(
                        false, a[p][kb], false, bt, (short)0, acc[p], false, false);
            }
#pragma unroll
            for (int p = 0; p < PTS; ++p)
#pragma unroll
                for (int r = 0; r < 8; ++r) {
                    float v = acc[p][r] + b1v[t];           // C: row r+8*half, col t*16+nidx
                    v = v > 0.f ? v : 0.f;
                    st[p * 2048 + (r + 8 * half) * 128 + (t * 16 + nidx)] = (_Float16)v;
                }
        }

        asm volatile("s_wait_dscnt 0" ::: "memory");  // same-wave LDS RAW fence

        // ---- re-layout C -> A for GEMM2 ----
        v16h a2[PTS][4];
#pragma unroll
        for (int p = 0; p < PTS; ++p)
#pragma unroll
            for (int kb = 0; kb < 4; ++kb) {
                const v8h lo = *(const v8h*)(st + p * 2048 + nidx * 128 + kb * 32 + K0);
                const v8h hi = *(const v8h*)(st + p * 2048 + nidx * 128 + kb * 32 + K0 + 8);
                v16h av;
#pragma unroll
                for (int i = 0; i < 8; ++i) { av[i] = lo[i]; av[i + 8] = hi[i]; }
                a2[p][kb] = av;
            }

        // ---- GEMM2 + b2, then max over 16 neighbor rows ----
#pragma unroll
        for (int t = 0; t < 8; ++t) {
            v8f acc[PTS];
#pragma unroll
            for (int p = 0; p < PTS; ++p) { v8f z = {}; acc[p] = z; }
#pragma unroll
            for (int kb = 0; kb < 4; ++kb) {
                const v16h bt = *(const v16h*)(
                    lw2 + ((t * 16 + nidx) * 128 + kb * 32 + half * 16));
#pragma unroll
                for (int p = 0; p < PTS; ++p)
                    acc[p] = __builtin_amdgcn_wmma_f32_16x16x32_f16(
                        false, a2[p][kb], false, bt, (short)0, acc[p], false, false);
            }
#pragma unroll
            for (int p = 0; p < PTS; ++p) {
                float m = acc[p][0];
#pragma unroll
                for (int r = 1; r < 8; ++r) m = fmaxf(m, acc[p][r]);
                // combine the two half-wave row groups (same column, lane ^ 16)
                const float o = __int_as_float(__builtin_amdgcn_ds_bpermute(
                    (ln ^ 16) << 2, __float_as_int(m)));
                m = fmaxf(m, o) + b2v[t];
                if (half == 0 && valid[p])
                    out[(long long)(base + p) * 128 + t * 16 + nidx] = m;
            }
        }
    }
}

extern "C" void kernel_launch(void* const* d_in, const int* in_sizes, int n_in,
                              void* d_out, int out_size, void* d_ws, size_t ws_size,
                              hipStream_t stream) {
    const float*     x   = (const float*)d_in[0];
    const long long* ind = (const long long*)d_in[1];   // int64 indices
    const float*     W1  = (const float*)d_in[2];
    const float*     b1  = (const float*)d_in[3];
    const float*     W2  = (const float*)d_in[4];
    const float*     b2  = (const float*)d_in[5];
    float*           out = (float*)d_out;

    const int npts = in_sizes[0] / 64;   // N from x's flat size (N*D)

    const int blocks = 1024;             // 4096 waves, ~12 point-pairs each
    edgeconv_wmma<<<blocks, WPB * 32, 0, stream>>>(x, ind, W1, b1, W2, b2, out, npts);
}
